// NattenFrameEncoder_13357348290800
// MI455X (gfx1250) — compile-verified, matrix operand order checked by reading, not run on hardware
//
#include <hip/hip_runtime.h>

typedef __bf16 bf16;
typedef __attribute__((ext_vector_type(8)))  __bf16 bf16x8;
typedef __attribute__((ext_vector_type(16))) __bf16 v16bf;
typedef __attribute__((ext_vector_type(8)))  float  v8f;

static constexpr int Lseq = 8192;
static constexpr int Fdim = 1024;
static constexpr int Cdim = 256;
static constexpr int Hn   = 8;
static constexpr int Dh   = 32;     // head dim
static constexpr int QKVS = 3 * Cdim;  // 768 row stride of qkv buffer
static constexpr int SPAN = 80;     // padded key span (window 63 + 15 tile skew, padded to 5*16)

__device__ __forceinline__ int imin(int a, int b) { return a < b ? a : b; }
__device__ __forceinline__ int imax(int a, int b) { return a > b ? a : b; }

__device__ __forceinline__ bf16x8 bzero8() {
  bf16x8 z;
#pragma unroll
  for (int i = 0; i < 8; ++i) z[i] = (bf16)0.0f;
  return z;
}

__device__ __forceinline__ v16bf pack16(bf16x8 lo, bf16x8 hi) {
  v16bf r;
#pragma unroll
  for (int i = 0; i < 8; ++i) { r[i] = lo[i]; r[i + 8] = hi[i]; }
  return r;
}

// Load a 16-bit WMMA A/B fragment: per-lane halves at [k0+kg*8, +8) and [k0+kg*8+16, +8)
__device__ __forceinline__ v16bf load_frag(const bf16* __restrict__ p, int kg) {
  bf16x8 lo = *(const bf16x8*)(p + kg * 8);
  bf16x8 hi = *(const bf16x8*)(p + kg * 8 + 16);
  return pack16(lo, hi);
}

// ---------------------------------------------------------------------------
// Generic bf16 WMMA GEMM:  C[M,N] = A[M,K] * BT[N,K]^T + bias, 1 wave = 16x64.
// ---------------------------------------------------------------------------
template <bool F32OUT>
__global__ __launch_bounds__(256) void wmma_gemm(const bf16* __restrict__ A,
                                                 const bf16* __restrict__ BT,
                                                 const float* __restrict__ bias,
                                                 void* __restrict__ Cout,
                                                 int M, int N, int K) {
  const int lane = threadIdx.x & 31;
  const int wid  = (blockIdx.x * blockDim.x + threadIdx.x) >> 5;
  const int ngrp = N >> 6;   // 64-col groups
  const int mt   = M >> 4;
  if (wid >= mt * ngrp) return;
  const int mtile = wid / ngrp;
  const int ng    = wid - mtile * ngrp;
  const int m0 = mtile << 4, n0 = ng << 6;
  const int m  = lane & 15,  kg = lane >> 4;

  v8f acc[4];
#pragma unroll
  for (int t = 0; t < 4; ++t) { v8f z = {}; acc[t] = z; }

  const bf16* arow = A + (size_t)(m0 + m) * K;
  const bf16* brow[4];
#pragma unroll
  for (int t = 0; t < 4; ++t) brow[t] = BT + (size_t)(n0 + t * 16 + m) * K;

  for (int k0 = 0; k0 < K; k0 += 32) {
    v16bf a = load_frag(arow + k0, kg);
#pragma unroll
    for (int t = 0; t < 4; ++t) {
      v16bf b = load_frag(brow[t] + k0, kg);
      acc[t] = __builtin_amdgcn_wmma_f32_16x16x32_bf16(false, a, false, b,
                                                       (short)0, acc[t], false, false);
    }
  }

#pragma unroll
  for (int t = 0; t < 4; ++t) {
    const int col = n0 + t * 16 + m;
    const float bv = bias[col];
#pragma unroll
    for (int r = 0; r < 8; ++r) {
      const int row = m0 + r + 8 * kg;
      const float v = acc[t][r] + bv;
      if (F32OUT) ((float*)Cout)[(size_t)row * N + col] = v;
      else        ((bf16*)Cout)[(size_t)row * N + col]  = (bf16)v;
    }
  }
}

// ---------------------------------------------------------------------------
// NATTEN-1D core: 1 wave = 16 queries x 1 head, key span padded to 80.
// ---------------------------------------------------------------------------
__global__ __launch_bounds__(256) void natten_wmma(const bf16* __restrict__ qkv,
                                                   bf16* __restrict__ aout) {
  __shared__ __align__(16) bf16 ldsP[8][16 * SPAN];  // P (probs), row-major 16x80 per wave
  __shared__ __align__(16) bf16 ldsV[8][Dh * SPAN];  // V transposed: [dim][key] per wave

  const int lane = threadIdx.x & 31;
  const int w    = threadIdx.x >> 5;
  const int job  = (blockIdx.x << 3) + w;   // (L/16)*H jobs
  const int h    = job & (Hn - 1);
  const int qt   = job >> 3;
  const int l0   = qt << 4;
  const int m = lane & 15, kg = lane >> 4;

  const int kb = imin(imax(l0 - 31, 0), Lseq - 63);  // base of padded key span

  // ---- S = (Q * scale) . K^T  (scale folded into scores) ----
  v16bf qa = load_frag(qkv + (size_t)(l0 + m) * QKVS + h * Dh, kg);
  v8f s[5];
#pragma unroll
  for (int t = 0; t < 5; ++t) {
    const int key = kb + t * 16 + m;
    v16bf kf;
    if (key < Lseq) kf = load_frag(qkv + (size_t)key * QKVS + Cdim + h * Dh, kg);
    else            kf = pack16(bzero8(), bzero8());
    v8f z = {};
    s[t] = __builtin_amdgcn_wmma_f32_16x16x32_bf16(false, qa, false, kf,
                                                   (short)0, z, false, false);
  }

  const float scale = 0.17677669529663687f;  // 1/sqrt(32)
  const float NEGI  = -1.0e30f;

  // ---- window mask ----
#pragma unroll
  for (int r = 0; r < 8; ++r) {
    const int l  = l0 + r + 8 * kg;
    const int st = imin(imax(l - 31, 0), Lseq - 63);
#pragma unroll
    for (int t = 0; t < 5; ++t) {
      const int j = kb + t * 16 + m;
      const float v = s[t][r] * scale;
      s[t][r] = (j >= st && j <= st + 62) ? v : NEGI;
    }
  }

  // ---- row softmax: reduce across 16 lanes of each half-wave + 5 tiles ----
#pragma unroll
  for (int r = 0; r < 8; ++r) {
    float mx = s[0][r];
#pragma unroll
    for (int t = 1; t < 5; ++t) mx = fmaxf(mx, s[t][r]);
    mx = fmaxf(mx, __shfl_xor(mx, 1, 32));
    mx = fmaxf(mx, __shfl_xor(mx, 2, 32));
    mx = fmaxf(mx, __shfl_xor(mx, 4, 32));
    mx = fmaxf(mx, __shfl_xor(mx, 8, 32));
    float sum = 0.0f;
#pragma unroll
    for (int t = 0; t < 5; ++t) { float p = __expf(s[t][r] - mx); s[t][r] = p; sum += p; }
    sum += __shfl_xor(sum, 1, 32);
    sum += __shfl_xor(sum, 2, 32);
    sum += __shfl_xor(sum, 4, 32);
    sum += __shfl_xor(sum, 8, 32);
    const float inv = 1.0f / sum;
#pragma unroll
    for (int t = 0; t < 5; ++t) s[t][r] *= inv;
  }

  // ---- stage P (C-layout -> row-major LDS) ----
  bf16* P = ldsP[w];
#pragma unroll
  for (int t = 0; t < 5; ++t) {
#pragma unroll
    for (int r = 0; r < 8; ++r) {
      P[(r + 8 * kg) * SPAN + t * 16 + m] = (bf16)s[t][r];
    }
  }

  // ---- stage V transposed: Vt[dim][key] ----
  bf16* Vt = ldsV[w];
  for (int jj = lane; jj < SPAN; jj += 32) {
    const int g = kb + jj;
    if (g < Lseq) {
      const bf16* vr = qkv + (size_t)g * QKVS + 2 * Cdim + h * Dh;
#pragma unroll
      for (int c8 = 0; c8 < 4; ++c8) {
        bf16x8 vv = *(const bf16x8*)(vr + c8 * 8);
#pragma unroll
        for (int e = 0; e < 8; ++e) Vt[(c8 * 8 + e) * SPAN + jj] = vv[e];
      }
    } else {
#pragma unroll
      for (int d = 0; d < Dh; ++d) Vt[d * SPAN + jj] = (bf16)0.0f;
    }
  }
  asm volatile("s_wait_dscnt 0" ::: "memory");   // LDS RAW fence (per-wave regions)

  // ---- O = P . V   (K padded 80 -> 96, N = 32 = two 16-col tiles) ----
  v8f o0 = {}, o1 = {};
#pragma unroll
  for (int kc = 0; kc < 3; ++kc) {
    const int cbase = kc * 32 + kg * 8;
    bf16x8 plo = *(const bf16x8*)(P + m * SPAN + cbase);
    bf16x8 phi = (cbase + 16 < SPAN) ? *(const bf16x8*)(P + m * SPAN + cbase + 16) : bzero8();
    v16bf pa = pack16(plo, phi);

    bf16x8 b0lo = *(const bf16x8*)(Vt + (0 * 16 + m) * SPAN + cbase);
    bf16x8 b0hi = (cbase + 16 < SPAN) ? *(const bf16x8*)(Vt + (0 * 16 + m) * SPAN + cbase + 16) : bzero8();
    o0 = __builtin_amdgcn_wmma_f32_16x16x32_bf16(false, pa, false, pack16(b0lo, b0hi),
                                                 (short)0, o0, false, false);

    bf16x8 b1lo = *(const bf16x8*)(Vt + (1 * 16 + m) * SPAN + cbase);
    bf16x8 b1hi = (cbase + 16 < SPAN) ? *(const bf16x8*)(Vt + (1 * 16 + m) * SPAN + cbase + 16) : bzero8();
    o1 = __builtin_amdgcn_wmma_f32_16x16x32_bf16(false, pa, false, pack16(b1lo, b1hi),
                                                 (short)0, o1, false, false);
  }

#pragma unroll
  for (int r = 0; r < 8; ++r) {
    const int l = l0 + r + 8 * kg;
    bf16* orow = aout + (size_t)l * Cdim + h * Dh;
    orow[m]      = (bf16)o0[r];
    orow[16 + m] = (bf16)o1[r];
  }
}

// ---------------------------------------------------------------------------
// fp32 -> bf16 conversion / transposed weight conversion
// ---------------------------------------------------------------------------
__global__ void f32_to_bf16(const float* __restrict__ in, bf16* __restrict__ out, int n) {
  int i = (blockIdx.x * blockDim.x + threadIdx.x) * 4;
  if (i + 3 < n) {
    float4 v = *(const float4*)(in + i);
    out[i] = (bf16)v.x; out[i + 1] = (bf16)v.y;
    out[i + 2] = (bf16)v.z; out[i + 3] = (bf16)v.w;
  } else {
    for (; i < n; ++i) out[i] = (bf16)in[i];
  }
}

__global__ void transpose_to_bf16(const float* __restrict__ W, bf16* __restrict__ WT,
                                  int K, int N) {  // W is K x N, WT is N x K
  int idx = blockIdx.x * blockDim.x + threadIdx.x;
  if (idx >= K * N) return;
  int k = idx / N, n = idx - k * N;
  WT[(size_t)n * K + k] = (bf16)W[idx];
}

// ---------------------------------------------------------------------------
extern "C" void kernel_launch(void* const* d_in, const int* in_sizes, int n_in,
                              void* d_out, int out_size, void* d_ws, size_t ws_size,
                              hipStream_t stream) {
  (void)in_sizes; (void)n_in; (void)out_size; (void)ws_size;
  const float* x    = (const float*)d_in[0];
  const float* Wp   = (const float*)d_in[1];
  const float* bp   = (const float*)d_in[2];
  const float* Wqkv = (const float*)d_in[3];
  const float* bqkv = (const float*)d_in[4];
  const float* Wo   = (const float*)d_in[5];
  const float* bo   = (const float*)d_in[6];
  float* out = (float*)d_out;

  char* ws = (char*)d_ws;
  size_t off = 0;
  auto wsalloc = [&](size_t bytes) -> void* {
    void* p = ws + off;
    off = (off + bytes + 255) & ~(size_t)255;
    return p;
  };
  bf16* xb    = (bf16*)wsalloc((size_t)Lseq * Fdim * 2);
  bf16* WpT   = (bf16*)wsalloc((size_t)Cdim * Fdim * 2);
  bf16* WqkvT = (bf16*)wsalloc((size_t)QKVS * Cdim * 2);
  bf16* WoT   = (bf16*)wsalloc((size_t)Cdim * Cdim * 2);
  bf16* y     = (bf16*)wsalloc((size_t)Lseq * Cdim * 2);
  bf16* qkvb  = (bf16*)wsalloc((size_t)Lseq * QKVS * 2);
  bf16* att   = (bf16*)wsalloc((size_t)Lseq * Cdim * 2);

  // Precision conversion + weight transposes (small; stays in L2)
  {
    int n = Lseq * Fdim;
    f32_to_bf16<<<(n / 4 + 255) / 256, 256, 0, stream>>>(x, xb, n);
  }
  {
    int n = Fdim * Cdim;
    transpose_to_bf16<<<(n + 255) / 256, 256, 0, stream>>>(Wp, WpT, Fdim, Cdim);
  }
  {
    int n = Cdim * QKVS;
    transpose_to_bf16<<<(n + 255) / 256, 256, 0, stream>>>(Wqkv, WqkvT, Cdim, QKVS);
  }
  {
    int n = Cdim * Cdim;
    transpose_to_bf16<<<(n + 255) / 256, 256, 0, stream>>>(Wo, WoT, Cdim, Cdim);
  }

  // y = x @ Wp + bp            (M=8192, N=256, K=1024)
  {
    int waves = (Lseq / 16) * (Cdim / 64);
    wmma_gemm<false><<<waves / 8, 256, 0, stream>>>(xb, WpT, bp, y, Lseq, Cdim, Fdim);
  }
  // qkv = y @ Wqkv + bqkv      (M=8192, N=768, K=256)
  {
    int waves = (Lseq / 16) * (QKVS / 64);
    wmma_gemm<false><<<waves / 8, 256, 0, stream>>>(y, WqkvT, bqkv, qkvb, Lseq, QKVS, Cdim);
  }
  // neighborhood attention     ((L/16)*H jobs, 1 wave each)
  {
    int jobs = (Lseq / 16) * Hn;
    natten_wmma<<<jobs / 8, 256, 0, stream>>>(qkvb, att);
  }
  // out = att @ Wo + bo        (M=8192, N=256, K=256) -> fp32
  {
    int waves = (Lseq / 16) * (Cdim / 64);
    wmma_gemm<true><<<waves / 8, 256, 0, stream>>>(att, WoT, bo, out, Lseq, Cdim, Cdim);
  }
}
